// MSAColumnAttention_71227737637047
// MI455X (gfx1250) — compile-verified, matrix operand order checked by reading, not run on hardware
//
#include <hip/hip_runtime.h>

typedef __attribute__((ext_vector_type(16))) _Float16 v16h;
typedef __attribute__((ext_vector_type(8)))  _Float16 v8h;
typedef __attribute__((ext_vector_type(8)))  float    v8f;
typedef __attribute__((ext_vector_type(4)))  float    v4f;

#define S_DIM 256
#define I_DIM 384
#define C_DIM 256
#define H_DIM 8
#define CH_DIM 32
#define ROWS  (S_DIM * I_DIM)   // 98304
#define LN_EPS 1e-5f

__device__ __forceinline__ v8f wmma_f16(v16h a, v16h b, v8f c) {
    // D = A(16x32) * B(32x16) + C, f32 accum
    return __builtin_amdgcn_wmma_f32_16x16x32_f16(
        false, a, false, b, (short)0, c, false, false);
}

// Load a 16-half A/B fragment: halves [0..7] from p, [8..15] from p+16.
// Matches the ISA 16-bit matrix VGPR layout when p = base + row*ld + kb,
// kb = (lane>>4)*8 (multiples of 8 halves -> 16B aligned -> b128 loads).
__device__ __forceinline__ v16h ld_frag(const _Float16* p) {
    v8h lo = *(const v8h*)p;
    v8h hi = *(const v8h*)(p + 16);
    v16h r;
#pragma unroll
    for (int j = 0; j < 8; ++j) { r[j] = lo[j]; r[8 + j] = hi[j]; }
    return r;
}

// ---------------- kernel 0: weight transpose + f32 -> f16 -------------------
__global__ void wt_kernel(const float* __restrict__ src, _Float16* __restrict__ dst,
                          int K, int N) {
    int idx = blockIdx.x * blockDim.x + threadIdx.x;
    if (idx < K * N) {
        int n = idx / K, k = idx % K;      // dst[n*K + k] = src[k*N + n]
        dst[idx] = (_Float16)src[(size_t)k * N + n];
    }
}

// ---------------- kernel 1: per-row LayerNorm stats -------------------------
__global__ __launch_bounds__(256) void ln_stats_kernel(const float* __restrict__ m,
                                                       float* __restrict__ stats) {
    int wave = threadIdx.x >> 5, lane = threadIdx.x & 31;
    int row  = blockIdx.x * 8 + wave;
    const float* p = m + (size_t)row * C_DIM + lane * 8;
    v4f a = *(const v4f*)p;
    v4f b = *(const v4f*)(p + 4);
    float s = 0.f, ss = 0.f;
#pragma unroll
    for (int j = 0; j < 4; ++j) { s += a[j] + b[j]; ss += a[j]*a[j] + b[j]*b[j]; }
#pragma unroll
    for (int d = 1; d < 32; d <<= 1) {
        s  += __shfl_xor(s,  d, 32);
        ss += __shfl_xor(ss, d, 32);
    }
    if (lane == 0) {
        float mu  = s * (1.0f / C_DIM);
        float var = ss * (1.0f / C_DIM) - mu * mu;
        stats[row * 2]     = mu;
        stats[row * 2 + 1] = rsqrtf(var + LN_EPS);
    }
}

// -------- kernel 2: fused LayerNorm + Q/K/V/Gate projection GEMM ------------
// One wave computes a 32x16 tile of all four 98304x256 outputs:
// 2 A fragments (LN'd rows) x 4 shared B fragments -> 8 WMMAs per K-step.
__global__ __launch_bounds__(256) void proj_kernel(
    const float* __restrict__ m, const float* __restrict__ stats,
    const float* __restrict__ gamma, const float* __restrict__ beta,
    const _Float16* __restrict__ wqT, const _Float16* __restrict__ wkT,
    const _Float16* __restrict__ wvT, const _Float16* __restrict__ wgT,
    const float* __restrict__ bg,
    _Float16* __restrict__ Yq, _Float16* __restrict__ Yk,
    _Float16* __restrict__ Yv, _Float16* __restrict__ Yg) {
    int wave = threadIdx.x >> 5, lane = threadIdx.x & 31;
    int task = blockIdx.x * 8 + wave;
    int mt = task >> 4, nt = task & 15;        // mt over 3072 32-row blocks
    int ml = lane & 15, kb = (lane >> 4) * 8;
    int row0 = mt * 32 + ml, row1 = row0 + 16;

    float mu0 = stats[row0 * 2], rs0 = stats[row0 * 2 + 1];
    float mu1 = stats[row1 * 2], rs1 = stats[row1 * 2 + 1];
    int ncol = nt * 16 + ml;

    v8f aq0 = {}, ak0 = {}, av0 = {}, ag0 = {};
    v8f aq1 = {}, ak1 = {}, av1 = {}, ag1 = {};
    const float* pm0 = m + (size_t)row0 * C_DIM;
    const float* pm1 = m + (size_t)row1 * C_DIM;
    for (int kk = 0; kk < C_DIM; kk += 32) {
        if (kk + 32 < C_DIM) {   // hint the next K-chunk (global_prefetch_b8)
            __builtin_prefetch(pm0 + kk + 32 + kb, 0, 0);
            __builtin_prefetch(pm1 + kk + 32 + kb, 0, 0);
        }
        v16h A0, A1;
#pragma unroll
        for (int j = 0; j < 8; ++j) {
            int c0 = kk + kb + j;
            int c1 = kk + kb + 16 + j;
            float g0 = gamma[c0], b0 = beta[c0];
            float g1 = gamma[c1], b1 = beta[c1];
            A0[j]     = (_Float16)((pm0[c0] - mu0) * rs0 * g0 + b0);
            A0[8 + j] = (_Float16)((pm0[c1] - mu0) * rs0 * g1 + b1);
            A1[j]     = (_Float16)((pm1[c0] - mu1) * rs1 * g0 + b0);
            A1[8 + j] = (_Float16)((pm1[c1] - mu1) * rs1 * g1 + b1);
        }
        size_t wo = (size_t)ncol * C_DIM + kk + kb;
        v16h Bq = ld_frag(wqT + wo);
        v16h Bk = ld_frag(wkT + wo);
        v16h Bv = ld_frag(wvT + wo);
        v16h Bg = ld_frag(wgT + wo);
        aq0 = wmma_f16(A0, Bq, aq0);  aq1 = wmma_f16(A1, Bq, aq1);
        ak0 = wmma_f16(A0, Bk, ak0);  ak1 = wmma_f16(A1, Bk, ak1);
        av0 = wmma_f16(A0, Bv, av0);  av1 = wmma_f16(A1, Bv, av1);
        ag0 = wmma_f16(A0, Bg, ag0);  ag1 = wmma_f16(A1, Bg, ag1);
    }
    int Mo = (lane >> 4) * 8;
    float bgn = bg[ncol];
#pragma unroll
    for (int r = 0; r < 8; ++r) {
        size_t i0 = (size_t)(mt * 32 + Mo + r) * C_DIM + ncol;
        size_t i1 = (size_t)(mt * 32 + 16 + Mo + r) * C_DIM + ncol;
        Yq[i0] = (_Float16)aq0[r];   Yq[i1] = (_Float16)aq1[r];
        Yk[i0] = (_Float16)ak0[r];   Yk[i1] = (_Float16)ak1[r];
        Yv[i0] = (_Float16)av0[r];   Yv[i1] = (_Float16)av1[r];
        Yg[i0] = (_Float16)(1.0f / (1.0f + __expf(-(ag0[r] + bgn))));
        Yg[i1] = (_Float16)(1.0f / (1.0f + __expf(-(ag1[r] + bgn))));
    }
}

// ---------------- kernel 3: column attention per (i, head) ------------------
__global__ __launch_bounds__(256) void attn_kernel(
    const _Float16* __restrict__ Yq, const _Float16* __restrict__ Yk,
    const _Float16* __restrict__ Yv, const _Float16* __restrict__ Yg,
    const float* __restrict__ mask, _Float16* __restrict__ Yo) {
    __shared__ _Float16 Vlin[S_DIM * CH_DIM];    // V rows (t, ch), async-staged: 16 KB
    __shared__ _Float16 VT[CH_DIM * S_DIM];      // V transposed (ch, t): 16 KB
    __shared__ _Float16 P[8][16 * S_DIM];        // per-wave prob strip: 64 KB

    int ih = blockIdx.x;
    int i = ih >> 3, h = ih & 7;
    int tid = threadIdx.x, wave = tid >> 5, lane = tid & 31;
    int ml = lane & 15, kb = (lane >> 4) * 8;

    // Async-stage this thread's 64B V row into LDS (ASYNCcnt-tracked), then
    // transpose LDS->LDS into VT[ch][t].
    {
        int t = tid;
        const _Float16* pv = Yv + ((size_t)(t * I_DIM + i)) * C_DIM + h * CH_DIM;
        unsigned int lds  = (unsigned int)(size_t)(&Vlin[t * CH_DIM]);
        unsigned long long ga = (unsigned long long)(size_t)pv;
        asm volatile(
            "global_load_async_to_lds_b128 %0, %1, off\n\t"
            "global_load_async_to_lds_b128 %0, %1, off offset:16\n\t"
            "global_load_async_to_lds_b128 %0, %1, off offset:32\n\t"
            "global_load_async_to_lds_b128 %0, %1, off offset:48"
            :: "v"(lds), "v"(ga) : "memory");
        asm volatile("s_wait_asynccnt 0x0" ::: "memory");
        const v8h* src = (const v8h*)(&Vlin[t * CH_DIM]);
#pragma unroll
        for (int c8 = 0; c8 < 4; ++c8) {
            v8h chunk = src[c8];
#pragma unroll
            for (int j = 0; j < 8; ++j) VT[(c8 * 8 + j) * S_DIM + t] = chunk[j];
        }
    }
    __syncthreads();

    // additive mask bias per lane per t-tile
    float biasl[16];
#pragma unroll
    for (int nt = 0; nt < 16; ++nt) {
        int t = nt * 16 + ml;
        biasl[nt] = 1e9f * (mask[(size_t)t * I_DIM + i] - 1.0f);
    }
    const float scale = 0.17677669529663687f;   // 1/sqrt(32)

    _Float16* Pw = P[wave];
    for (int mt = 0; mt < 2; ++mt) {
        int s0 = wave * 32 + mt * 16;
        // Q fragment: rows s0..s0+15, full K = ch = 32 (one WMMA per score tile)
        const _Float16* pq = Yq + ((size_t)((s0 + ml) * I_DIM + i)) * C_DIM + h * CH_DIM;
        v16h Aq = ld_frag(pq + kb);

        v8f acc[16];
#pragma unroll
        for (int nt = 0; nt < 16; ++nt) {
            int t0 = nt * 16 + ml;
            const _Float16* pk = Yk + ((size_t)(t0 * I_DIM + i)) * C_DIM + h * CH_DIM;
            v16h Bk = ld_frag(pk + kb);
            v8f z = {};
            acc[nt] = wmma_f16(Aq, Bk, z);
        }

        // softmax over t (rows striped across the 16-lane halves in C-layout)
        float rinv[8];
#pragma unroll
        for (int r = 0; r < 8; ++r) {
            float mx = -1e30f;
#pragma unroll
            for (int nt = 0; nt < 16; ++nt) {
                float v = acc[nt][r] * scale + biasl[nt];
                acc[nt][r] = v;
                mx = fmaxf(mx, v);
            }
#pragma unroll
            for (int d = 1; d < 16; d <<= 1) mx = fmaxf(mx, __shfl_xor(mx, d, 32));
            float sum = 0.f;
#pragma unroll
            for (int nt = 0; nt < 16; ++nt) {
                float e = __expf(acc[nt][r] - mx);
                acc[nt][r] = e;
                sum += e;
            }
#pragma unroll
            for (int d = 1; d < 16; d <<= 1) sum += __shfl_xor(sum, d, 32);
            rinv[r] = 1.0f / sum;
        }

        // write probabilities (f16) to this wave's LDS strip
#pragma unroll
        for (int r = 0; r < 8; ++r) {
            int M = (lane >> 4) * 8 + r;
#pragma unroll
            for (int nt = 0; nt < 16; ++nt)
                Pw[M * S_DIM + nt * 16 + ml] = (_Float16)(acc[nt][r] * rinv[r]);
        }

        // O = P(16x256) x V(256x32): 8 K-steps, 2 ch tiles
        v8f oa0 = {}, oa1 = {};
        for (int kk = 0; kk < S_DIM; kk += 32) {
            v16h Ap = ld_frag(Pw + ml * S_DIM + kk + kb);
            v16h B0 = ld_frag(VT + (0 * 16 + ml) * S_DIM + kk + kb);
            v16h B1 = ld_frag(VT + (1 * 16 + ml) * S_DIM + kk + kb);
            oa0 = wmma_f16(Ap, B0, oa0);
            oa1 = wmma_f16(Ap, B1, oa1);
        }

        // gate and store
#pragma unroll
        for (int r = 0; r < 8; ++r) {
            int s = s0 + (lane >> 4) * 8 + r;
            size_t base = ((size_t)(s * I_DIM + i)) * C_DIM + h * CH_DIM;
            size_t i0 = base + ml;
            size_t i1 = base + 16 + ml;
            Yo[i0] = (_Float16)((float)Yg[i0] * oa0[r]);
            Yo[i1] = (_Float16)((float)Yg[i1] * oa1[r]);
        }
    }
}

// ---------------- kernel 4: output projection + bias (f32) ------------------
// One wave computes a 32x16 tile: B fragment reused by 2 WMMAs per K-step.
__global__ __launch_bounds__(256) void out_kernel(
    const _Float16* __restrict__ Yo, const _Float16* __restrict__ woT,
    const float* __restrict__ bo, float* __restrict__ out) {
    int wave = threadIdx.x >> 5, lane = threadIdx.x & 31;
    int task = blockIdx.x * 8 + wave;
    int mt = task >> 4, nt = task & 15;
    int ml = lane & 15, kb = (lane >> 4) * 8;
    int row0 = mt * 32 + ml, row1 = row0 + 16;
    int ncol = nt * 16 + ml;

    v8f acc0 = {}, acc1 = {};
    for (int kk = 0; kk < C_DIM; kk += 32) {
        v16h A0 = ld_frag(Yo + (size_t)row0 * C_DIM + kk + kb);
        v16h A1 = ld_frag(Yo + (size_t)row1 * C_DIM + kk + kb);
        v16h B  = ld_frag(woT + (size_t)ncol * C_DIM + kk + kb);
        acc0 = wmma_f16(A0, B, acc0);
        acc1 = wmma_f16(A1, B, acc1);
    }
    int Mo = (lane >> 4) * 8;
    float bias = bo[ncol];
#pragma unroll
    for (int r = 0; r < 8; ++r) {
        out[(size_t)(mt * 32 + Mo + r) * C_DIM + ncol]      = acc0[r] + bias;
        out[(size_t)(mt * 32 + 16 + Mo + r) * C_DIM + ncol] = acc1[r] + bias;
    }
}

// ---------------------------------------------------------------------------
extern "C" void kernel_launch(void* const* d_in, const int* in_sizes, int n_in,
                              void* d_out, int out_size, void* d_ws, size_t ws_size,
                              hipStream_t stream) {
    const float* m     = (const float*)d_in[0];
    const float* mask  = (const float*)d_in[1];
    const float* gamma = (const float*)d_in[2];
    const float* beta  = (const float*)d_in[3];
    const float* Wq    = (const float*)d_in[4];
    const float* Wk    = (const float*)d_in[5];
    const float* Wv    = (const float*)d_in[6];
    const float* Wg    = (const float*)d_in[7];
    const float* bg    = (const float*)d_in[8];
    const float* Wo    = (const float*)d_in[9];
    const float* bo    = (const float*)d_in[10];
    float* out = (float*)d_out;

    char* ws = (char*)d_ws;
    const size_t WSZ = (size_t)C_DIM * C_DIM * sizeof(_Float16);  // 128 KB
    const size_t YSZ = (size_t)ROWS * C_DIM * sizeof(_Float16);   // 48 MB
    _Float16* wqT = (_Float16*)ws;           ws += WSZ;
    _Float16* wkT = (_Float16*)ws;           ws += WSZ;
    _Float16* wvT = (_Float16*)ws;           ws += WSZ;
    _Float16* wgT = (_Float16*)ws;           ws += WSZ;
    _Float16* woT = (_Float16*)ws;           ws += WSZ;
    float*    stats = (float*)ws;            ws += (size_t)ROWS * 2 * sizeof(float);
    _Float16* Yq = (_Float16*)ws;            ws += YSZ;
    _Float16* Yk = (_Float16*)ws;            ws += YSZ;
    _Float16* Yv = (_Float16*)ws;            ws += YSZ;
    _Float16* Yg = (_Float16*)ws;            ws += YSZ;
    _Float16* Yo = (_Float16*)ws;            ws += YSZ;

    // 0) weight prep
    int wtElems = C_DIM * C_DIM;
    dim3 wtGrid((wtElems + 255) / 256), tb(256);
    wt_kernel<<<wtGrid, tb, 0, stream>>>(Wq, wqT, C_DIM, C_DIM);
    wt_kernel<<<wtGrid, tb, 0, stream>>>(Wk, wkT, C_DIM, C_DIM);
    wt_kernel<<<wtGrid, tb, 0, stream>>>(Wv, wvT, C_DIM, C_DIM);
    wt_kernel<<<wtGrid, tb, 0, stream>>>(Wg, wgT, C_DIM, C_DIM);
    wt_kernel<<<wtGrid, tb, 0, stream>>>(Wo, woT, C_DIM, C_DIM);

    // 1) LayerNorm stats (8 rows per block, one wave per row)
    ln_stats_kernel<<<dim3(ROWS / 8), tb, 0, stream>>>(m, stats);

    // 2) fused LN + 4-way projection: 3072 32-row tiles x 16 N-tiles, 8 waves/block
    proj_kernel<<<dim3((ROWS / 32) * 16 / 8), tb, 0, stream>>>(
        m, stats, gamma, beta, wqT, wkT, wvT, wgT, bg, Yq, Yk, Yv, Yg);

    // 3) attention: one block per (i, head)
    attn_kernel<<<dim3(I_DIM * H_DIM), tb, 0, stream>>>(Yq, Yk, Yv, Yg, mask, Yo);

    // 4) output projection
    out_kernel<<<dim3((ROWS / 32) * 16 / 8), tb, 0, stream>>>(Yo, woT, bo, out);
}
